// attention_13340168421794
// MI455X (gfx1250) — matrix operand order repaired and checked by reading, not compiled, not run on hardware
//
#include <hip/hip_runtime.h>
#include <hip/hip_bf16.h>

typedef _Float16 h16;
typedef __attribute__((ext_vector_type(16))) _Float16 v16h;
typedef __attribute__((ext_vector_type(8)))  float    v8f;

#define DIMS   1024
#define HEADS  16
#define HD     64
#define ROWS   16384          // 4 * 4096 global rows
#define WIN    512
#define NWI    8              // convergence groups (nwin)
#define NB     4
#define GRP_ELEMS (NB*HEADS*WIN*HD)   // 2097152 elements per group

// ---------------------------------------------------------------------------
// WMMA helpers (CDNA5 v_wmma_f32_16x16x32_f16, wave32)
// ---------------------------------------------------------------------------
__device__ __forceinline__ v8f wmma16(v16h a, v16h b, v8f c) {
  return __builtin_amdgcn_wmma_f32_16x16x32_f16(false, a, false, b, (short)0, c,
                                                false, false);
}

// A fragment (16x32 f16, row-major source, ld = lda). ISA layout:
// lane<16 : m=lane,   K = {hi*8..hi*8+7, 16+hi*8..}, hi=0
// lane>=16: m=lane-16, same with hi=1
__device__ __forceinline__ v16h load_a_frag(const h16* A, int lda, int m0,
                                            int kb, int lane) {
  int m  = m0 + (lane & 15);
  int hi = lane >> 4;
  const h16* p = A + (size_t)m * lda + kb + hi * 8;
  v16h a;
#pragma unroll
  for (int i = 0; i < 8; ++i) a[i] = p[i];
#pragma unroll
  for (int i = 0; i < 8; ++i) a[8 + i] = p[16 + i];
  return a;
}

// B fragment (32x16): B[k][n] = W[n][k] with W row-major [N,K].
// lane n = lane&15, holds K = {hi*8..hi*8+7, 16+hi*8..16+hi*8+7}, hi = lane>=16.
__device__ __forceinline__ v16h load_b_frag(const h16* W, int ldb, int n0,
                                            int kb, int lane) {
  int n = n0 + (lane & 15);
  const h16* p = W + (size_t)n * ldb + kb + (lane >> 4) * 8;
  v16h b;
#pragma unroll
  for (int i = 0; i < 8; ++i) { b[i] = p[i]; b[8 + i] = p[16 + i]; }
  return b;
}

// ---------------------------------------------------------------------------
// LayerNorm over 1024-wide rows (lna), output f16
// ---------------------------------------------------------------------------
__global__ __launch_bounds__(256) void ln_row_kernel(const float* __restrict__ x,
                                                     const float* __restrict__ w,
                                                     h16* __restrict__ out) {
  __shared__ float red[8];
  int row = blockIdx.x;
  const float* xr = x + (size_t)row * DIMS;
  float v[4];
#pragma unroll
  for (int j = 0; j < 4; ++j) v[j] = xr[threadIdx.x + j * 256];
  float s = v[0] + v[1] + v[2] + v[3];
#pragma unroll
  for (int o = 16; o >= 1; o >>= 1) s += __shfl_xor(s, o, 32);
  if ((threadIdx.x & 31) == 0) red[threadIdx.x >> 5] = s;
  __syncthreads();
  float tot = 0.f;
#pragma unroll
  for (int j = 0; j < 8; ++j) tot += red[j];
  float mean = tot * (1.0f / DIMS);
  float var = 0.f;
#pragma unroll
  for (int j = 0; j < 4; ++j) { v[j] -= mean; var += v[j] * v[j]; }
#pragma unroll
  for (int o = 16; o >= 1; o >>= 1) var += __shfl_xor(var, o, 32);
  __syncthreads();
  if ((threadIdx.x & 31) == 0) red[threadIdx.x >> 5] = var;
  __syncthreads();
  float vtot = 0.f;
#pragma unroll
  for (int j = 0; j < 8; ++j) vtot += red[j];
  float rinv = rsqrtf(vtot * (1.0f / DIMS) + 1e-5f);
  h16* orow = out + (size_t)row * DIMS;
#pragma unroll
  for (int j = 0; j < 4; ++j)
    orow[threadIdx.x + j * 256] =
        (h16)(v[j] * rinv * w[threadIdx.x + j * 256]);
}

__global__ void cast_f16_kernel(const float* __restrict__ a,
                                h16* __restrict__ b, int n) {
  int i = blockIdx.x * 256 + threadIdx.x;
  if (i < n) b[i] = (h16)a[i];
}

// ---------------------------------------------------------------------------
// Generic WMMA GEMM: C[M,N] = A_h[M,K] * W_h[N,K]^T (+ bias[n])
// Register-blocked: each wave computes a 32x64 tile (2 A-frags x 4 B-frags,
// 8 independent accumulators -> 8 WMMAs per k-step, no D->A/B RAW hazard).
// 8 waves/block -> 256 rows x 64 cols per block.
// ---------------------------------------------------------------------------
__global__ __launch_bounds__(256) void gemm_f16_kernel(
    const h16* __restrict__ A, const h16* __restrict__ W,
    const float* __restrict__ bias, float* __restrict__ C, int M, int N, int K) {
  int lane = threadIdx.x & 31;
  int wave = threadIdx.x >> 5;
  int n0 = blockIdx.x * 64;
  int m0 = (blockIdx.y * 8 + wave) * 32;
  if (m0 >= M || n0 >= N) return;
  v8f acc[2][4];
#pragma unroll
  for (int mi = 0; mi < 2; ++mi)
#pragma unroll
    for (int t = 0; t < 4; ++t) acc[mi][t] = (v8f){};

  for (int kb = 0; kb < K; kb += 32) {
    v16h a0 = load_a_frag(A, K, m0, kb, lane);
    v16h a1 = load_a_frag(A, K, m0 + 16, kb, lane);
#pragma unroll
    for (int t = 0; t < 4; ++t) {
      v16h b = load_b_frag(W, K, n0 + t * 16, kb, lane);
      acc[0][t] = wmma16(a0, b, acc[0][t]);
      acc[1][t] = wmma16(a1, b, acc[1][t]);
    }
  }
  int hi = lane >> 4, nl = lane & 15;
#pragma unroll
  for (int t = 0; t < 4; ++t) {
    int n = n0 + t * 16 + nl;
    float bv = bias ? bias[n] : 0.0f;
#pragma unroll
    for (int mi = 0; mi < 2; ++mi) {
      int mb = m0 + mi * 16 + hi * 8;
#pragma unroll
      for (int r = 0; r < 8; ++r)
        C[(size_t)(mb + r) * N + n] = acc[mi][t][r] + bv;
    }
  }
}

// ---------------------------------------------------------------------------
// Post-QKV: q=(Q+bq)*s -> head-LN -> qcur(f32); k=K*s -> head-LN -> kh(f16);
// v=V+bv -> vh(f16). Also zeroes prev (which aliases Vraw: read V first!).
// One wave per (row g, head h); head-dim=64 -> 2 elems/lane.
// ---------------------------------------------------------------------------
__global__ __launch_bounds__(256) void postqkv_kernel(
    const float* __restrict__ Q, const float* __restrict__ Kr,
    const float* __restrict__ V, const float* __restrict__ bq,
    const float* __restrict__ bv, const float* __restrict__ lnb,
    float* __restrict__ qcur, h16* __restrict__ kh, h16* __restrict__ vh,
    float* __restrict__ prev) {
  int idx = blockIdx.x * 8 + (threadIdx.x >> 5);
  if (idx >= ROWS * HEADS) return;
  int lane = threadIdx.x & 31;
  int g = idx >> 4, h = idx & 15;
  size_t off = (size_t)g * DIMS + h * 64;
  const float scale = 0.3535533905932738f;  // 64^-0.25

  // read V first (prev aliases V storage)
  float v0 = V[off + lane] + bv[h * 64 + lane];
  float v1 = V[off + 32 + lane] + bv[h * 64 + 32 + lane];

  // ---- q ----
  float a0 = (Q[off + lane] + bq[h * 64 + lane]) * scale;
  float a1 = (Q[off + 32 + lane] + bq[h * 64 + 32 + lane]) * scale;
  float s = a0 + a1;
#pragma unroll
  for (int o = 16; o >= 1; o >>= 1) s += __shfl_xor(s, o, 32);
  float mean = s * (1.0f / 64);
  a0 -= mean; a1 -= mean;
  float var = a0 * a0 + a1 * a1;
#pragma unroll
  for (int o = 16; o >= 1; o >>= 1) var += __shfl_xor(var, o, 32);
  float rinv = rsqrtf(var * (1.0f / 64) + 1e-5f);
  qcur[off + lane]      = a0 * rinv * lnb[lane];
  qcur[off + 32 + lane] = a1 * rinv * lnb[32 + lane];

  // ---- k ----
  a0 = Kr[off + lane] * scale;
  a1 = Kr[off + 32 + lane] * scale;
  s = a0 + a1;
#pragma unroll
  for (int o = 16; o >= 1; o >>= 1) s += __shfl_xor(s, o, 32);
  mean = s * (1.0f / 64);
  a0 -= mean; a1 -= mean;
  var = a0 * a0 + a1 * a1;
#pragma unroll
  for (int o = 16; o >= 1; o >>= 1) var += __shfl_xor(var, o, 32);
  rinv = rsqrtf(var * (1.0f / 64) + 1e-5f);
  kh[off + lane]      = (h16)(a0 * rinv * lnb[lane]);
  kh[off + 32 + lane] = (h16)(a1 * rinv * lnb[32 + lane]);

  // ---- v + zero prev ----
  vh[off + lane]        = (h16)v0;
  vh[off + 32 + lane]   = (h16)v1;
  prev[off + lane]      = 0.0f;
  prev[off + 32 + lane] = 0.0f;
}

// ---------------------------------------------------------------------------
// Small per-head projection (64x64 weight) via WMMA, optional LN epilogue,
// optional transposed output ([w][h][64][512] for vpT).
// One wave per (16-row tile, head).
// ---------------------------------------------------------------------------
template <bool IN_F32, bool DO_LN, bool TOUT>
__global__ __launch_bounds__(256) void proj64_kernel(
    const void* __restrict__ In, const h16* __restrict__ Wsm,
    const float* __restrict__ bias, const float* __restrict__ lnb,
    const int* __restrict__ done, h16* __restrict__ Out) {
  int task = blockIdx.x * 8 + (threadIdx.x >> 5);
  if (task >= (ROWS / 16) * HEADS) return;
  int lane = threadIdx.x & 31;
  int h  = task & 15;
  int m0 = (task >> 4) * 16;
  if (done) {
    int nwi = (m0 & 4095) >> 9;
    if (done[nwi]) return;
  }
  int hi = lane >> 4, nl = lane & 15;

  v16h a0, a1;
  if (IN_F32) {
    const float* A = (const float*)In;
    const float* p = A + (size_t)(m0 + nl) * DIMS + h * 64;
#pragma unroll
    for (int i = 0; i < 8; ++i) {
      a0[i]     = (h16)p[hi * 8 + i];
      a0[8 + i] = (h16)p[16 + hi * 8 + i];
      a1[i]     = (h16)p[32 + hi * 8 + i];
      a1[8 + i] = (h16)p[48 + hi * 8 + i];
    }
  } else {
    const h16* A = (const h16*)In + h * 64;
    a0 = load_a_frag(A, DIMS, m0, 0, lane);
    a1 = load_a_frag(A, DIMS, m0, 32, lane);
  }

  v8f acc[4];
#pragma unroll
  for (int t = 0; t < 4; ++t) {
    v16h b0 = load_b_frag(Wsm, 64, t * 16, 0, lane);
    v16h b1 = load_b_frag(Wsm, 64, t * 16, 32, lane);
    v8f c = {};
    c = wmma16(a0, b0, c);
    c = wmma16(a1, b1, c);
    acc[t] = c;
  }
#pragma unroll
  for (int t = 0; t < 4; ++t) {
    float bb = bias[t * 16 + nl];
#pragma unroll
    for (int r = 0; r < 8; ++r) acc[t][r] += bb;
  }
  if constexpr (DO_LN) {
#pragma unroll
    for (int r = 0; r < 8; ++r) {
      float s = acc[0][r] + acc[1][r] + acc[2][r] + acc[3][r];
#pragma unroll
      for (int o = 8; o >= 1; o >>= 1) s += __shfl_xor(s, o, 32);
      float mean = s * (1.0f / 64);
      float d0 = acc[0][r] - mean, d1 = acc[1][r] - mean;
      float d2 = acc[2][r] - mean, d3 = acc[3][r] - mean;
      float var = d0 * d0 + d1 * d1 + d2 * d2 + d3 * d3;
#pragma unroll
      for (int o = 8; o >= 1; o >>= 1) var += __shfl_xor(var, o, 32);
      float rinv = rsqrtf(var * (1.0f / 64) + 1e-5f);
      acc[0][r] = d0 * rinv * lnb[nl];
      acc[1][r] = d1 * rinv * lnb[16 + nl];
      acc[2][r] = d2 * rinv * lnb[32 + nl];
      acc[3][r] = d3 * rinv * lnb[48 + nl];
    }
  }
  if constexpr (!TOUT) {
    int mb = m0 + hi * 8;
#pragma unroll
    for (int t = 0; t < 4; ++t)
#pragma unroll
      for (int r = 0; r < 8; ++r)
        Out[(size_t)(mb + r) * DIMS + h * 64 + t * 16 + nl] = (h16)acc[t][r];
  } else {
    int bi = m0 >> 12, nwi = (m0 & 4095) >> 9, s0 = m0 & 511;
    size_t base = (((size_t)(bi * 8 + nwi)) * 16 + h) * (size_t)(64 * 512);
    int mb = s0 + hi * 8;
#pragma unroll
    for (int t = 0; t < 4; ++t)
#pragma unroll
      for (int r = 0; r < 8; ++r)
        Out[base + (size_t)(t * 16 + nl) * 512 + mb + r] = (h16)acc[t][r];
  }
}

// ---------------------------------------------------------------------------
// Causal flash attention per (window, head, 16-query tile), online softmax.
// s = qp kp^T * 1/8 ; o = softmax(s) vp ; keys streamed 32 at a time.
// p transposed to A-fragment layout through per-wave LDS (s_wait_dscnt).
// ---------------------------------------------------------------------------
__global__ __launch_bounds__(256) void attn_kernel(
    const h16* __restrict__ qp, const h16* __restrict__ kp,
    const h16* __restrict__ vpT, float* __restrict__ O,
    const int* __restrict__ done) {
  __shared__ h16 plds[8][16 * 32];
  int task = blockIdx.x * 8 + (threadIdx.x >> 5);
  if (task >= 32 * HEADS * (WIN / 16)) return;
  int lane = threadIdx.x & 31;
  int wave = threadIdx.x >> 5;
  int qt = task & 31;
  int h  = (task >> 5) & 15;
  int w  = task >> 9;                 // w = bi*8 + nwi
  int bi = w >> 3, nwi = w & 7;
  if (done[nwi]) return;
  int gbase = bi * 4096 + nwi * 512;
  int m0 = qt * 16;
  int hi = lane >> 4, nl = lane & 15;

  const h16* Ap = qp + h * 64;
  v16h aq0 = load_a_frag(Ap, DIMS, gbase + m0, 0, lane);
  v16h aq1 = load_a_frag(Ap, DIMS, gbase + m0, 32, lane);
  const h16* Kp = kp + h * 64;
  size_t vbase = (((size_t)w) * 16 + h) * (size_t)(64 * 512);

  float M[8], L[8];
  v8f acc[4] = {{}, {}, {}, {}};
#pragma unroll
  for (int r = 0; r < 8; ++r) { M[r] = -1e30f; L[r] = 0.f; }

  h16* lp = plds[wave];
  int nch = (m0 + 16 + 31) >> 5;  // causal: keys 0 .. m0+15
  for (int ch = 0; ch < nch; ++ch) {
    int k0 = ch * 32;
    // ---- scores: two 16-key tiles ----
    v8f st[2];
#pragma unroll
    for (int t = 0; t < 2; ++t) {
      v16h b0 = load_b_frag(Kp, DIMS, gbase + k0 + t * 16, 0, lane);
      v16h b1 = load_b_frag(Kp, DIMS, gbase + k0 + t * 16, 32, lane);
      v8f c = {};
      c = wmma16(aq0, b0, c);
      c = wmma16(aq1, b1, c);
      st[t] = c;
    }
    // ---- scale + causal mask + chunk row-max, online softmax state ----
    float alpha[8];
#pragma unroll
    for (int r = 0; r < 8; ++r) {
      int qi = m0 + hi * 8 + r;
      float mx = -1e30f;
#pragma unroll
      for (int t = 0; t < 2; ++t) {
        int kj = k0 + t * 16 + nl;
        float vv = st[t][r] * 0.125f;     // 1/sqrt(64)
        if (kj > qi) vv = -1e30f;
        st[t][r] = vv;
        mx = fmaxf(mx, vv);
      }
#pragma unroll
      for (int o = 8; o >= 1; o >>= 1) mx = fmaxf(mx, __shfl_xor(mx, o, 32));
      float nm = fmaxf(M[r], mx);
      alpha[r] = __expf(M[r] - nm);
      M[r] = nm;
      float rs = 0.f;
#pragma unroll
      for (int t = 0; t < 2; ++t) {
        float p = __expf(st[t][r] - nm);
        st[t][r] = p;
        rs += p;
      }
#pragma unroll
      for (int o = 8; o >= 1; o >>= 1) rs += __shfl_xor(rs, o, 32);
      L[r] = L[r] * alpha[r] + rs;
    }
    // ---- transpose p into A-fragment layout via LDS ----
#pragma unroll
    for (int t = 0; t < 2; ++t)
#pragma unroll
      for (int r = 0; r < 8; ++r)
        lp[(hi * 8 + r) * 32 + t * 16 + nl] = (h16)st[t][r];
    asm volatile("s_wait_dscnt 0" ::: "memory");
    v16h ap;
#pragma unroll
    for (int i = 0; i < 8; ++i) {
      ap[i]     = lp[nl * 32 + hi * 8 + i];
      ap[8 + i] = lp[nl * 32 + 16 + hi * 8 + i];
    }
    // ---- rescale and accumulate o += p @ vp ----
#pragma unroll
    for (int t = 0; t < 4; ++t) {
#pragma unroll
      for (int r = 0; r < 8; ++r) acc[t][r] *= alpha[r];
      const h16* vp = vpT + vbase + (size_t)(t * 16 + nl) * 512 + k0 + hi * 8;
      v16h bv;
#pragma unroll
      for (int i = 0; i < 8; ++i) { bv[i] = vp[i]; bv[8 + i] = vp[16 + i]; }
      acc[t] = wmma16(ap, bv, acc[t]);
    }
  }
  // ---- epilogue ----
  int mb = gbase + m0 + hi * 8;
#pragma unroll
  for (int t = 0; t < 4; ++t)
#pragma unroll
    for (int r = 0; r < 8; ++r)
      O[(size_t)(mb + r) * DIMS + h * 64 + t * 16 + nl] = acc[t][r] / L[r];
}

// ---------------------------------------------------------------------------
// Convergence machinery (per nwin group)
// ---------------------------------------------------------------------------
__global__ void initflags_kernel(int* done, int* conv, float* diffacc) {
  int i = threadIdx.x;
  if (i < 8) { done[i] = 0; conv[i] = 0; }
  if (i < 24) diffacc[i] = 0.f;
}

__global__ __launch_bounds__(256) void diff_kernel(
    const float* __restrict__ O, const float* __restrict__ prev,
    float* __restrict__ diffacc, const int* __restrict__ done) {
  const int BPG = GRP_ELEMS / 2048;  // 1024 blocks per group
  int grp = blockIdx.x / BPG;
  if (done[grp]) return;
  size_t e0 = ((size_t)(blockIdx.x % BPG) * 256 + threadIdx.x) * 8;
  float accv = 0.f;
#pragma unroll
  for (int j = 0; j < 8; ++j) {
    size_t e = e0 + j;
    int d2 = (int)(e & 63);
    size_t r = e >> 6;
    int s2 = (int)(r & 511); r >>= 9;
    int h2 = (int)(r & 15);
    int bi = (int)(r >> 4);
    size_t g1 = (size_t)bi * 4096 + grp * 512 + (h2 * 32 + (s2 >> 4));
    float val = O[g1 * DIMS + (s2 & 15) * 64 + d2];
    size_t g2 = (size_t)bi * 4096 + grp * 512 + s2;
    accv += fabsf(val - prev[g2 * DIMS + h2 * 64 + d2]);
  }
#pragma unroll
  for (int o = 16; o >= 1; o >>= 1) accv += __shfl_xor(accv, o, 32);
  __shared__ float red[8];
  if ((threadIdx.x & 31) == 0) red[threadIdx.x >> 5] = accv;
  __syncthreads();
  if (threadIdx.x == 0) {
    float t = 0.f;
    for (int j = 0; j < 8; ++j) t += red[j];
    atomicAdd(diffacc + grp, t);
  }
}

__global__ void conv_kernel(const float* diffacc, int* conv, int it) {
  int i = threadIdx.x;
  if (i < 8) {
    float d = diffacc[i] * (1.0f / (float)GRP_ELEMS);
    conv[i] = ((it > 0) && (d < 0.01f + 0.1f * d)) ? 1 : 0;
  }
}

__global__ __launch_bounds__(256) void update_kernel(
    const float* __restrict__ O, float* __restrict__ attn,
    float* __restrict__ qcur, float* __restrict__ prev,
    const int* __restrict__ done, const int* __restrict__ conv) {
  const int BPG = GRP_ELEMS / 2048;
  int grp = blockIdx.x / BPG;
  if (done[grp]) return;          // done: attn/qcur/prev untouched
  int upd = conv[grp];            // done==0 here, so upd = conv
  size_t e0 = ((size_t)(blockIdx.x % BPG) * 256 + threadIdx.x) * 8;
#pragma unroll
  for (int j = 0; j < 8; ++j) {
    size_t e = e0 + j;
    int d2 = (int)(e & 63);
    size_t r = e >> 6;
    int s2 = (int)(r & 511); r >>= 9;
    int h2 = (int)(r & 15);
    int bi = (int)(r >> 4);
    size_t g1 = (size_t)bi * 4096 + grp * 512 + (h2 * 32 + (s2 >> 4));
    float val = O[g1 * DIMS + (s2 & 15) * 64 + d2];
    size_t o2 = ((size_t)bi * 4096 + grp * 512 + s2) * DIMS + h2 * 64 + d2;
    attn[o2] = val;               // attn_out = where(done, attn, iter_out)
    if (!upd) {                   // upd = done||conv
      qcur[o2] += val;
      prev[o2] = val;
    }
  }
}

__global__ void doneup_kernel(int* done, const int* conv) {
  int i = threadIdx.x;
  if (i < 8) done[i] |= conv[i];
}

// ---------------------------------------------------------------------------
extern "C" void kernel_launch(void* const* d_in, const int* in_sizes, int n_in,
                              void* d_out, int out_size, void* d_ws,
                              size_t ws_size, hipStream_t stream) {
  const float* x   = (const float*)d_in[0];
  const float* Wq  = (const float*)d_in[1];
  const float* bq  = (const float*)d_in[2];
  const float* Wk  = (const float*)d_in[3];
  const float* Wv  = (const float*)d_in[4];
  const float* bv  = (const float*)d_in[5];
  const float* Wo  = (const float*)d_in[6];
  const float* bo  = (const float*)d_in[7];
  const float* lna = (const float*)d_in[8];
  const float* lnb = (const float*)d_in[9];
  const float* lqw = (const float*)d_in[10];
  const float* lqb = (const float*)d_in[11];
  const float* lkw = (const float*)d_in[12];
  const float* lkb = (const float*)d_in[13];
  const float* lvw = (const float*)d_in[14];
  const float* lvb = (const float*)d_in[15];
  (void)in_sizes; (void)n_in; (void)out_size;

  char* ws = (char*)d_ws;
  size_t off = 0;
  auto alloc = [&](size_t bytes) -> char* {
    char* p = ws + off;
    off += (bytes + 255) & ~(size_t)255;
    return p;
  };
  h16*   xn   = (h16*)alloc((size_t)ROWS * DIMS * 2);   // -> attn_h later
  h16*   wqh  = (h16*)alloc((size_t)DIMS * DIMS * 2);
  h16*   wkh  = (h16*)alloc((size_t)DIMS * DIMS * 2);
  h16*   wvh  = (h16*)alloc((size_t)DIMS * DIMS * 2);
  h16*   woh  = (h16*)alloc((size_t)DIMS * DIMS * 2);
  h16*   lqh  = (h16*)alloc(64 * 64 * 2);
  h16*   lkh  = (h16*)alloc(64 * 64 * 2);
  h16*   lvh  = (h16*)alloc(64 * 64 * 2);
  float* qraw = (float*)alloc((size_t)ROWS * DIMS * 4); // -> O later
  float* kraw = (float*)alloc((size_t)ROWS * DIMS * 4); // -> attn later
  float* vraw = (float*)alloc((size_t)ROWS * DIMS * 4); // -> prev later
  float* qcur = (float*)alloc((size_t)ROWS * DIMS * 4);
  h16*   kh   = (h16*)alloc((size_t)ROWS * DIMS * 2);
  h16*   vh   = (h16*)alloc((size_t)ROWS * DIMS * 2);
  h16*   kp   = (h16*)alloc((size_t)ROWS * DIMS * 2);
  h16*   vpT  = (h16*)alloc((size_t)ROWS * DIMS * 2);
  h16*   qp   = (h16*)alloc((size_t)ROWS * DIMS * 2);
  float* diffacc = (float*)alloc(24 * 4);
  int*   done = (int*)alloc(8 * 4);
  int*   conv = (int*)alloc(8 * 4);
  if (off > ws_size) return;  // insufficient scratch

  float* Obuf   = qraw;  // Qraw dead after postqkv
  float* attn   = kraw;  // Kraw dead after postqkv
  float* prev   = vraw;  // Vraw dead after postqkv (read-before-write in-kernel)
  h16*   attn_h = xn;    // xn dead after the three big GEMMs

  initflags_kernel<<<1, 64, 0, stream>>>(done, conv, diffacc);
  ln_row_kernel<<<ROWS, 256, 0, stream>>>(x, lna, xn);
  cast_f16_kernel<<<(DIMS * DIMS + 255) / 256, 256, 0, stream>>>(Wq, wqh, DIMS * DIMS);
  cast_f16_kernel<<<(DIMS * DIMS + 255) / 256, 256, 0, stream>>>(Wk, wkh, DIMS * DIMS);
  cast_f16_kernel<<<(DIMS * DIMS + 255) / 256, 256, 0, stream>>>(Wv, wvh, DIMS * DIMS);
  cast_f16_kernel<<<(DIMS * DIMS + 255) / 256, 256, 0, stream>>>(Wo, woh, DIMS * DIMS);
  cast_f16_kernel<<<16, 256, 0, stream>>>(lqw, lqh, 64 * 64);
  cast_f16_kernel<<<16, 256, 0, stream>>>(lkw, lkh, 64 * 64);
  cast_f16_kernel<<<16, 256, 0, stream>>>(lvw, lvh, 64 * 64);

  // 32x64 tile per wave: grid = (N/64, M/256)
  dim3 gg(DIMS / 64, ROWS / 256);
  gemm_f16_kernel<<<gg, 256, 0, stream>>>(xn, wqh, nullptr, qraw, ROWS, DIMS, DIMS);
  gemm_f16_kernel<<<gg, 256, 0, stream>>>(xn, wkh, nullptr, kraw, ROWS, DIMS, DIMS);
  gemm_f16_kernel<<<gg, 256, 0, stream>>>(xn, wvh, nullptr, vraw, ROWS, DIMS, DIMS);

  postqkv_kernel<<<ROWS * HEADS / 8, 256, 0, stream>>>(qraw, kraw, vraw, bq, bv,
                                                       lnb, qcur, kh, vh, prev);
  // loop-invariant small projections
  proj64_kernel<false, true, false><<<2048, 256, 0, stream>>>(kh, lkh, lkb, lnb, nullptr, kp);
  proj64_kernel<false, false, true><<<2048, 256, 0, stream>>>(vh, lvh, lvb, nullptr, nullptr, vpT);

  for (int it = 0; it < 3; ++it) {
    proj64_kernel<true, true, false><<<2048, 256, 0, stream>>>(qcur, lqh, lqb, lnb, done, qp);
    attn_kernel<<<2048, 256, 0, stream>>>(qp, kp, vpT, Obuf, done);
    diff_kernel<<<NWI * (GRP_ELEMS / 2048), 256, 0, stream>>>(Obuf, prev, diffacc + it * 8, done);
    conv_kernel<<<1, 32, 0, stream>>>(diffacc + it * 8, conv, it);
    update_kernel<<<NWI * (GRP_ELEMS / 2048), 256, 0, stream>>>(Obuf, attn, qcur, prev, done, conv);
    doneup_kernel<<<1, 32, 0, stream>>>(done, conv);
  }

  cast_f16_kernel<<<(ROWS * DIMS + 255) / 256, 256, 0, stream>>>(attn, attn_h, ROWS * DIMS);
  gemm_f16_kernel<<<gg, 256, 0, stream>>>(attn_h, woh, bo, (float*)d_out, ROWS, DIMS, DIMS);
}
